// BridgeTowerBlock_57595511439397
// MI455X (gfx1250) — matrix operand order repaired and checked by reading, not compiled, not run on hardware
//
#include <hip/hip_runtime.h>
#include <hip/hip_bf16.h>

typedef __attribute__((ext_vector_type(16))) __bf16 v16bf;
typedef __attribute__((ext_vector_type(8)))  float  v8f;
typedef unsigned int v4u __attribute__((ext_vector_type(4)));
typedef int          v4i __attribute__((ext_vector_type(4)));
typedef int          v8i __attribute__((ext_vector_type(8)));

#define HDIM 512
#define FFNH 1024
typedef __hip_bfloat16 bf;

// ---------------------------------------------------------------------------
// Weight convert + transpose: W[K x Nout] fp32 -> Wt[Nout x K] bf16
// ---------------------------------------------------------------------------
__global__ void k_conv_wt(const float* __restrict__ W, bf* __restrict__ Wt,
                          int K, int Nout) {
  int idx = blockIdx.x * blockDim.x + threadIdx.x;
  int total = K * Nout;
  if (idx >= total) return;
  int n = idx / K;
  int k = idx - n * K;
  Wt[idx] = __float2bfloat16(W[(size_t)k * Nout + n]);
}

// ---------------------------------------------------------------------------
// Activation convert: fp32 -> bf16 (vectorized by 4)
// ---------------------------------------------------------------------------
__global__ void k_conv_bf(const float* __restrict__ X, bf* __restrict__ Y, int total) {
  int idx = (blockIdx.x * blockDim.x + threadIdx.x) * 4;
  if (idx >= total) return;
  float4 v = *reinterpret_cast<const float4*>(X + idx);
  union { bf h[4]; uint2 u; } o;
  o.h[0] = __float2bfloat16(v.x);
  o.h[1] = __float2bfloat16(v.y);
  o.h[2] = __float2bfloat16(v.z);
  o.h[3] = __float2bfloat16(v.w);
  *reinterpret_cast<uint2*>(Y + idx) = o.u;
}

// ---------------------------------------------------------------------------
// Fragment loader: two 16B chunks -> v16bf
// ---------------------------------------------------------------------------
__device__ __forceinline__ v16bf frag2(const void* p0, const void* p1) {
  union { v16bf v; uint4 u[2]; } f;
  f.u[0] = *reinterpret_cast<const uint4*>(p0);
  f.u[1] = *reinterpret_cast<const uint4*>(p1);
  return f.v;
}

// ---------------------------------------------------------------------------
// TDM: DMA a [rows x K] bf16 tile (row-major, row stride K) from global to LDS.
// D# layout per CDNA5 ISA 8.3-8.6; 2D tile, data_size=2B, type=2.
// ---------------------------------------------------------------------------
__device__ __forceinline__ void tdm_load_tile(unsigned lds_off, const bf* gptr,
                                              int K, int rows) {
  unsigned long long ga = (unsigned long long)gptr;
  v4u g0;
  g0[0] = 1u;                                              // count=1, user mode
  g0[1] = lds_off;                                         // lds_addr (bytes)
  g0[2] = (unsigned)(ga & 0xFFFFFFFFu);                    // global_addr[31:0]
  g0[3] = (unsigned)((ga >> 32) & 0x01FFFFFFu) | (2u << 30); // addr[56:32] | type=2
  v8i g1;
  g1[0] = (int)(1u << 16);                                 // data_size=1 (2 bytes)
  g1[1] = (int)(((unsigned)K & 0xFFFFu) << 16);            // tensor_dim0[15:0] @ bits 48..63
  g1[2] = (int)((((unsigned)K >> 16) & 0xFFFFu) |          // tensor_dim0[31:16]
                (((unsigned)rows & 0xFFFFu) << 16));       // tensor_dim1[15:0]
  g1[3] = (int)(((unsigned)K & 0xFFFFu) << 16);            // tensor_dim1 hi=0 | tile_dim0=K
  g1[4] = rows;                                            // tile_dim1=rows, tile_dim2=0
  g1[5] = K;                                               // tensor_dim0_stride[31:0] = K
  g1[6] = 0;
  g1[7] = 0;
  v4i z4 = {0, 0, 0, 0};
#if __clang_major__ >= 23
  v8i z8 = {0, 0, 0, 0, 0, 0, 0, 0};
  __builtin_amdgcn_tensor_load_to_lds(g0, g1, z4, z4, z8, 0);
#else
  __builtin_amdgcn_tensor_load_to_lds(g0, g1, z4, z4, 0);
#endif
}

// ---------------------------------------------------------------------------
// bf16 WMMA GEMM (generic epilogue): C[M x Nout] = A[M x K] @ Wt[Nout x K]^T
// Tile 128x64 per WG (8 waves), 32x32 per wave. Used for bridge + FFN-up.
// ---------------------------------------------------------------------------
__global__ __launch_bounds__(256)
void k_gemm(const bf* __restrict__ A, const bf* __restrict__ Wt,
            const float* __restrict__ bias,
            const float* __restrict__ res1, const float* __restrict__ res2,
            float* __restrict__ outF, bf* __restrict__ outB,
            int Nout, int K, int relu)
{
  const int lane = threadIdx.x & 31;
  const int wave = threadIdx.x >> 5;
  const int m0 = blockIdx.y * 128 + (wave >> 1) * 32;
  const int n0 = blockIdx.x * 64  + (wave & 1) * 32;

  const int arow0 = m0 + (lane & 15);
  const int akb   = (lane >> 4) * 8;
  const int bcol0 = n0 + (lane & 15);
  const int bkb   = (lane >> 4) * 8;

  const bf* pa0 = A  + (size_t)arow0        * K + akb;
  const bf* pa1 = A  + (size_t)(arow0 + 16) * K + akb;
  const bf* pb0 = Wt + (size_t)bcol0        * K + bkb;
  const bf* pb1 = Wt + (size_t)(bcol0 + 16) * K + bkb;

  v8f acc00 = {}, acc01 = {}, acc10 = {}, acc11 = {};
  for (int k0 = 0; k0 < K; k0 += 32) {
    v16bf a0 = frag2(pa0 + k0, pa0 + k0 + 16);
    v16bf a1 = frag2(pa1 + k0, pa1 + k0 + 16);
    v16bf b0 = frag2(pb0 + k0, pb0 + k0 + 16);
    v16bf b1 = frag2(pb1 + k0, pb1 + k0 + 16);
    acc00 = __builtin_amdgcn_wmma_f32_16x16x32_bf16(false, a0, false, b0, (short)0, acc00, false, false);
    acc01 = __builtin_amdgcn_wmma_f32_16x16x32_bf16(false, a0, false, b1, (short)0, acc01, false, false);
    acc10 = __builtin_amdgcn_wmma_f32_16x16x32_bf16(false, a1, false, b0, (short)0, acc10, false, false);
    acc11 = __builtin_amdgcn_wmma_f32_16x16x32_bf16(false, a1, false, b1, (short)0, acc11, false, false);
  }

  v8f accs[2][2] = { { acc00, acc01 }, { acc10, acc11 } };
  const int cl   = lane & 15;
  const int rOff = (lane >> 4) * 8;
#pragma unroll
  for (int sm = 0; sm < 2; ++sm)
#pragma unroll
    for (int sn = 0; sn < 2; ++sn) {
      const int col = n0 + sn * 16 + cl;
      const float bb = bias ? bias[col] : 0.0f;
#pragma unroll
      for (int i = 0; i < 8; ++i) {
        const int row = m0 + sm * 16 + rOff + i;
        const size_t off = (size_t)row * Nout + col;
        float c = accs[sm][sn][i] + bb;
        if (relu) c = fmaxf(c, 0.0f);
        if (res1) c += res1[off];
        if (res2) c += res2[off];
        if (outF) outF[off] = c;
        if (outB) outB[off] = __float2bfloat16(c);
      }
    }
}

// ---------------------------------------------------------------------------
// Fused GEMM + bias + relu + residual(s) + LayerNorm.  Nout = 512 (full row).
// WG tile: 32 rows x 512 cols; wave w owns cols w*64..w*64+63 (2x4 subtiles).
// A tile (32 x K bf16) is DMA'd to LDS once by the TDM; K-loop reads A from
// LDS (ds_load_b128) and streams L2-hot transposed weights from global.
// Row stats: shfl-reduce within 16-lane groups, LDS reduce across 8 waves.
// ---------------------------------------------------------------------------
__global__ __launch_bounds__(256)
void k_gemm_ln(const bf* __restrict__ A, const bf* __restrict__ Wt,
               const float* __restrict__ bias,
               const float* __restrict__ res1, const float* __restrict__ res2,
               const float* __restrict__ lng, const float* __restrict__ lnb,
               float* __restrict__ outF, bf* __restrict__ outB,
               int K, int relu)
{
  extern __shared__ char lds_raw[];
  const int lane = threadIdx.x & 31;
  const int wave = threadIdx.x >> 5;
  const int m0 = blockIdx.x * 32;
  const int n0 = wave * 64;
  const unsigned aBytes = 32u * (unsigned)K * 2u;

  // --- stage A tile via Tensor Data Mover ---------------------------------
  if (wave == 0) {
    tdm_load_tile(__builtin_amdgcn_groupstaticsize(), A + (size_t)m0 * K, K, 32);
    __builtin_amdgcn_s_wait_tensorcnt(0);
  }
  __syncthreads();

  const int cl   = lane & 15;
  const int akb  = (lane >> 4) * 8;
  const int bkb  = (lane >> 4) * 8;
  const char* lA0 = lds_raw + ((unsigned)cl        * K + akb) * 2;  // rows sm=0
  const char* lA1 = lds_raw + ((unsigned)(cl + 16) * K + akb) * 2;  // rows sm=1
  const bf* pb[4];
#pragma unroll
  for (int sn = 0; sn < 4; ++sn)
    pb[sn] = Wt + (size_t)(n0 + sn * 16 + cl) * K + bkb;

  v8f acc[2][4] = {};
  for (int k0 = 0; k0 < K; k0 += 32) {
    v16bf a0 = frag2(lA0 + k0 * 2, lA0 + k0 * 2 + 32);
    v16bf a1 = frag2(lA1 + k0 * 2, lA1 + k0 * 2 + 32);
#pragma unroll
    for (int sn = 0; sn < 4; ++sn) {
      v16bf b = frag2(pb[sn] + k0, pb[sn] + k0 + 16);
      acc[0][sn] = __builtin_amdgcn_wmma_f32_16x16x32_bf16(false, a0, false, b, (short)0, acc[0][sn], false, false);
      acc[1][sn] = __builtin_amdgcn_wmma_f32_16x16x32_bf16(false, a1, false, b, (short)0, acc[1][sn], false, false);
    }
  }

  // --- epilogue: bias / relu / residuals ----------------------------------
  const int rOff = (lane >> 4) * 8;
  float bbias[4], gg[4], bbeta[4];
#pragma unroll
  for (int sn = 0; sn < 4; ++sn) {
    const int col = n0 + sn * 16 + cl;
    bbias[sn] = bias ? bias[col] : 0.0f;
    gg[sn] = lng[col];
    bbeta[sn] = lnb[col];
  }
#pragma unroll
  for (int sm = 0; sm < 2; ++sm)
#pragma unroll
    for (int sn = 0; sn < 4; ++sn)
#pragma unroll
      for (int i = 0; i < 8; ++i) {
        const int row = m0 + sm * 16 + rOff + i;
        const size_t off = (size_t)row * HDIM + n0 + sn * 16 + cl;
        float c = acc[sm][sn][i] + bbias[sn];
        if (relu) c = fmaxf(c, 0.0f);
        if (res1) c += res1[off];
        if (res2) c += res2[off];
        acc[sm][sn][i] = c;
      }

  // --- row statistics: 64-col partial per wave, then cross-wave via LDS ---
  float* redS = (float*)(lds_raw + aBytes);  // [8 waves][32 rows]
  float* redQ = redS + 256;
#pragma unroll
  for (int sm = 0; sm < 2; ++sm)
#pragma unroll
    for (int i = 0; i < 8; ++i) {
      float s = acc[sm][0][i] + acc[sm][1][i] + acc[sm][2][i] + acc[sm][3][i];
      float q = acc[sm][0][i] * acc[sm][0][i] + acc[sm][1][i] * acc[sm][1][i]
              + acc[sm][2][i] * acc[sm][2][i] + acc[sm][3][i] * acc[sm][3][i];
#pragma unroll
      for (int msk = 1; msk <= 8; msk <<= 1) {
        s += __shfl_xor(s, msk, 32);
        q += __shfl_xor(q, msk, 32);
      }
      if (cl == 0) {
        const int rl = sm * 16 + rOff + i;
        redS[wave * 32 + rl] = s;
        redQ[wave * 32 + rl] = q;
      }
    }
  __syncthreads();

  // --- normalize + store --------------------------------------------------
#pragma unroll
  for (int sm = 0; sm < 2; ++sm)
#pragma unroll
    for (int i = 0; i < 8; ++i) {
      const int rl = sm * 16 + rOff + i;
      float s = 0.0f, q = 0.0f;
#pragma unroll
      for (int w = 0; w < 8; ++w) { s += redS[w * 32 + rl]; q += redQ[w * 32 + rl]; }
      const float mu  = s * (1.0f / HDIM);
      const float var = q * (1.0f / HDIM) - mu * mu;
      const float rs  = rsqrtf(var + 1e-5f);
      const int row = m0 + rl;
#pragma unroll
      for (int sn = 0; sn < 4; ++sn) {
        const size_t off = (size_t)row * HDIM + n0 + sn * 16 + cl;
        const float y = (acc[sm][sn][i] - mu) * rs * gg[sn] + bbeta[sn];
        if (outF) outF[off] = y;
        if (outB) outB[off] = __float2bfloat16(y);
      }
    }
}

// ---------------------------------------------------------------------------
extern "C" void kernel_launch(void* const* d_in, const int* in_sizes, int n_in,
                              void* d_out, int out_size, void* d_ws, size_t ws_size,
                              hipStream_t stream) {
  const int N = in_sizes[0] / HDIM;   // 65536 rows per modality

  const float* m1 = (const float*)d_in[0];
  const float* m2 = (const float*)d_in[1];
  const float* bw1    = (const float*)d_in[2];  const float* bb1    = (const float*)d_in[3];
  const float* bw2    = (const float*)d_in[4];  const float* bb2    = (const float*)d_in[5];
  const float* sa1_wv = (const float*)d_in[10]; const float* sa1_bv = (const float*)d_in[11];
  const float* sa2_wv = (const float*)d_in[16]; const float* sa2_bv = (const float*)d_in[17];
  const float* ca1_wv = (const float*)d_in[22]; const float* ca1_bv = (const float*)d_in[23];
  const float* ca2_wv = (const float*)d_in[28]; const float* ca2_bv = (const float*)d_in[29];
  const float* f1w1 = (const float*)d_in[30]; const float* f1b1 = (const float*)d_in[31];
  const float* f1w2 = (const float*)d_in[32]; const float* f1b2 = (const float*)d_in[33];
  const float* f2w1 = (const float*)d_in[34]; const float* f2b1 = (const float*)d_in[35];
  const float* f2w2 = (const float*)d_in[36]; const float* f2b2 = (const float*)d_in[37];
  const float* ln1g = (const float*)d_in[38]; const float* ln1b = (const float*)d_in[39];
  const float* ln2g = (const float*)d_in[40]; const float* ln2b = (const float*)d_in[41];
  const float* ln3g = (const float*)d_in[42]; const float* ln3b = (const float*)d_in[43];
  const float* ln4g = (const float*)d_in[44]; const float* ln4b = (const float*)d_in[45];
  const float* ln5g = (const float*)d_in[46]; const float* ln5b = (const float*)d_in[47];
  const float* ln6g = (const float*)d_in[48]; const float* ln6b = (const float*)d_in[49];

  float* out1 = (float*)d_out;
  float* out2 = out1 + (size_t)N * HDIM;

  char* p = (char*)d_ws;
  auto carve = [&](size_t bytes) -> char* {
    char* r = p;
    p += (bytes + 255) & ~(size_t)255;
    return r;
  };
  const size_t WW = (size_t)HDIM * HDIM;
  const size_t WF = (size_t)HDIM * FFNH;
  bf* wt_b1  = (bf*)carve(WW * 2);  bf* wt_b2  = (bf*)carve(WW * 2);
  bf* wt_sv1 = (bf*)carve(WW * 2);  bf* wt_sv2 = (bf*)carve(WW * 2);
  bf* wt_cv1 = (bf*)carve(WW * 2);  bf* wt_cv2 = (bf*)carve(WW * 2);
  bf* wt_f1a = (bf*)carve(WF * 2);  bf* wt_f1b = (bf*)carve(WF * 2);
  bf* wt_f2a = (bf*)carve(WF * 2);  bf* wt_f2b = (bf*)carve(WF * 2);
  const size_t NH = (size_t)N * HDIM;
  float* F1 = (float*)carve(NH * 4);        // O1 -> X1 -> Y1 (fp32 state, modality 1)
  float* F3 = (float*)carve(NH * 4);        // O2 -> X2 -> Y2 (fp32 state, modality 2)
  bf* mb1 = (bf*)carve(NH * 2);             // m1 bf16 -> X1 bf16
  bf* mb2 = (bf*)carve(NH * 2);             // m2 bf16 -> X2 bf16 -> Y2 bf16
  bf* Obf = (bf*)carve(NH * 2);             // bridge bf16 -> Y1 bf16
  bf* Hbf = (bf*)carve((size_t)N * FFNH * 2);

  const dim3 blk(256);
  const int gWW = (int)((WW + 255) / 256), gWF = (int)((WF + 255) / 256);
  k_conv_wt<<<gWW, blk, 0, stream>>>(bw1,    wt_b1,  HDIM, HDIM);
  k_conv_wt<<<gWW, blk, 0, stream>>>(bw2,    wt_b2,  HDIM, HDIM);
  k_conv_wt<<<gWW, blk, 0, stream>>>(sa1_wv, wt_sv1, HDIM, HDIM);
  k_conv_wt<<<gWW, blk, 0, stream>>>(sa2_wv, wt_sv2, HDIM, HDIM);
  k_conv_wt<<<gWW, blk, 0, stream>>>(ca1_wv, wt_cv1, HDIM, HDIM);
  k_conv_wt<<<gWW, blk, 0, stream>>>(ca2_wv, wt_cv2, HDIM, HDIM);
  k_conv_wt<<<gWF, blk, 0, stream>>>(f1w1, wt_f1a, HDIM, FFNH);
  k_conv_wt<<<gWF, blk, 0, stream>>>(f1w2, wt_f1b, FFNH, HDIM);
  k_conv_wt<<<gWF, blk, 0, stream>>>(f2w1, wt_f2a, HDIM, FFNH);
  k_conv_wt<<<gWF, blk, 0, stream>>>(f2w2, wt_f2b, FFNH, HDIM);
  const int gAC = (int)((NH / 4 + 255) / 256);
  k_conv_bf<<<gAC, blk, 0, stream>>>(m1, mb1, (int)NH);
  k_conv_bf<<<gAC, blk, 0, stream>>>(m2, mb2, (int)NH);

  const dim3 g512(HDIM / 64, N / 128);
  const dim3 g1024(FFNH / 64, N / 128);
  const int gFR = N / 32;                                  // full-row fused grid
  const size_t sh512  = 32u * 512u  * 2u + 2048u;
  const size_t sh1024 = 32u * 1024u * 2u + 2048u;

  // ---- stage A: bridge + self-attn(v) + LN (fused) -------------------------
  k_gemm   <<<g512, blk, 0,      stream>>>(mb1, wt_b1, bb1, nullptr, nullptr, F1, Obf, HDIM, HDIM, 0);
  k_gemm_ln<<<gFR,  blk, sh512,  stream>>>(Obf, wt_sv1, sa1_bv, F1, m1, ln1g, ln1b, F1, mb1, HDIM, 1);
  k_gemm   <<<g512, blk, 0,      stream>>>(mb2, wt_b2, bb2, nullptr, nullptr, F3, Obf, HDIM, HDIM, 0);
  k_gemm_ln<<<gFR,  blk, sh512,  stream>>>(Obf, wt_sv2, sa2_bv, F3, m2, ln2g, ln2b, F3, mb2, HDIM, 1);

  // ---- stage B: cross-attn(v) + LN (fused); u = relu(x_other@Wv+b) + 2x ---
  k_gemm_ln<<<gFR,  blk, sh512,  stream>>>(mb2, wt_cv1, ca1_bv, F1, F1, ln3g, ln3b, F1, Obf, HDIM, 1);
  k_gemm_ln<<<gFR,  blk, sh512,  stream>>>(mb1, wt_cv2, ca2_bv, F3, F3, ln4g, ln4b, F3, mb2, HDIM, 1);

  // ---- stage C: FFN + LN (fused down-proj), modality 1 ---------------------
  k_gemm   <<<g1024, blk, 0,      stream>>>(Obf, wt_f1a, f1b1, nullptr, nullptr, nullptr, Hbf, FFNH, HDIM, 1);
  k_gemm_ln<<<gFR,   blk, sh1024, stream>>>(Hbf, wt_f1b, f1b2, F1, nullptr, ln5g, ln5b, out1, nullptr, FFNH, 0);

  // ---- stage C: FFN + LN (fused down-proj), modality 2 ---------------------
  k_gemm   <<<g1024, blk, 0,      stream>>>(mb2, wt_f2a, f2b1, nullptr, nullptr, nullptr, Hbf, FFNH, HDIM, 1);
  k_gemm_ln<<<gFR,   blk, sh1024, stream>>>(Hbf, wt_f2b, f2b2, F3, nullptr, ln6g, ln6b, out2, nullptr, FFNH, 0);
}
